// DAC_block_80324478370162
// MI455X (gfx1250) — compile-verified, matrix operand order checked by reading, not run on hardware
//
#include <hip/hip_runtime.h>
#include <hip/hip_bf16.h>

#define D_MODEL 256
#define D_INNER 512
#define D_STATE 32
#define D_CONV  4
#define DT_RANK 16
#define H_MLP   32
#define BINS    256
#define BB      16
#define LL      2048
#define NTOK    (BB*LL)   // 32768 tokens

typedef __bf16 bf16;
typedef __attribute__((ext_vector_type(16))) __bf16 v16bf;
typedef __attribute__((ext_vector_type(8)))  __bf16 v8bf;
typedef __attribute__((ext_vector_type(8)))  float  v8f;
typedef __attribute__((ext_vector_type(4)))  int    v4i;

// CDNA5 async global->LDS path (guarded; falls back to plain LDS staging)
#if defined(__HIP_DEVICE_COMPILE__) && \
    __has_builtin(__builtin_amdgcn_global_load_async_to_lds_b128) && \
    __has_builtin(__builtin_amdgcn_s_wait_asynccnt)
#define USE_ASYNC_LDS 1
#else
#define USE_ASYNC_LDS 0
#endif

#define AS_GLOBAL __attribute__((address_space(1)))
#define AS_SHARED __attribute__((address_space(3)))

// ---------------------------------------------------------------------------
// f32 -> bf16 elementwise convert (activations + weights)
// ---------------------------------------------------------------------------
__global__ void cvt_bf16_kernel(const float* __restrict__ in, bf16* __restrict__ out, int n) {
    int i = blockIdx.x * blockDim.x + threadIdx.x;
    if (i < n) out[i] = (bf16)in[i];
}

// ---------------------------------------------------------------------------
// WMMA bf16 GEMM: C[M,N] = epi(A[M,K] @ B[N,K]^T + bias)
// block = 128 threads (4 waves); wave w owns rows [blockIdx.x*64 + w*16, +16).
// All 4 waves share the same NT 16-wide N-tiles; the B chunk for each 32-K
// step is staged into LDS (async, double-buffered) once per block.
// Geometry contract: N == gridDim.y * NT * 16, K % 32 == 0, M % 64 == 0.
// EPI: 0 = none, f32 store | 1 = none, bf16 store
//      2 = leaky_relu(0.01)+bias, bf16 store | 3 = sigmoid+bias, f32 store
// ---------------------------------------------------------------------------
union ABpack { v16bf v; v8bf h[2]; };

template <int EPI, int NT>
__global__ void wmma_gemm_kernel(const bf16* __restrict__ A,
                                 const bf16* __restrict__ Bm,
                                 const float* __restrict__ bias,
                                 void* __restrict__ Cout,
                                 int M, int N, int K)
{
    constexpr int COLS  = NT * 16;            // output columns per block
    constexpr int UNITS = COLS * 4;           // 16B load units per 32-K stage
    constexpr int TRIPS = (UNITS + 127) / 128;// async instrs per thread per stage

    __shared__ __align__(16) bf16 sB[2][COLS * 32];

    const int tid  = threadIdx.x;
    const int lane = tid & 31;
    const int wave = tid >> 5;
    const int m0   = (blockIdx.x * 4 + wave) * 16;
    const int n0b  = blockIdx.y * COLS;
    const int lo   = lane & 15;
    const int hi8  = (lane >> 4) * 8;         // K-half select (ISA 16-bit A/B layout)

    const bf16* arow = A + (size_t)(m0 + lo) * K;

    v8f acc[NT];
    #pragma unroll
    for (int t = 0; t < NT; ++t) acc[t] = 0.0f;

    // Stage the 32-K-deep B chunk for k0 into sB[bufsel].
    // Every thread issues exactly TRIPS loads (wrapped), so ASYNCcnt bookkeeping
    // is wave-uniform and s_wait_asynccnt(TRIPS) is a valid double-buffer fence.
    auto stage = [&](int k0, int bufsel) {
        #pragma unroll
        for (int i = 0; i < TRIPS; ++i) {
            int j   = (tid + i * 128) % UNITS;
            int col = j >> 2;                 // 0..COLS-1
            int seg = j & 3;                  // 8-elem (16B) segment within 32-K row
            const bf16* gp = Bm + (size_t)(n0b + col) * K + k0 + seg * 8;
            bf16*       lp = &sB[bufsel][col * 32 + seg * 8];
#if USE_ASYNC_LDS
            __builtin_amdgcn_global_load_async_to_lds_b128(
                (AS_GLOBAL v4i*)gp, (AS_SHARED v4i*)lp, 0, 0);
#else
            *(uint4*)lp = *(const uint4*)gp;
#endif
        }
    };

    stage(0, 0);
    int buf = 0;
    for (int k0 = 0; k0 < K; k0 += 32) {
        // A fragment: direct global->VGPR (unique per wave)
        ABpack a;
        a.h[0] = *(const v8bf*)(arow + k0 + hi8);
        a.h[1] = *(const v8bf*)(arow + k0 + hi8 + 16);

        if (k0 + 32 < K) {
            stage(k0 + 32, buf ^ 1);          // prefetch next stage
#if USE_ASYNC_LDS
            __builtin_amdgcn_s_wait_asynccnt(TRIPS);   // current stage landed
#endif
        } else {
#if USE_ASYNC_LDS
            __builtin_amdgcn_s_wait_asynccnt(0);
#endif
        }
        __syncthreads();                      // stage visible to all waves

        #pragma unroll
        for (int t = 0; t < NT; ++t) {
            const bf16* bp = &sB[buf][(t * 16 + lo) * 32];
            ABpack b;
            b.h[0] = *(const v8bf*)(bp + hi8);
            b.h[1] = *(const v8bf*)(bp + hi8 + 16);
            acc[t] = __builtin_amdgcn_wmma_f32_16x16x32_bf16(
                         false, a.v, false, b.v, (short)0, acc[t], false, false);
        }
        __syncthreads();                      // all reads done before buf is re-staged
        buf ^= 1;
    }

    #pragma unroll
    for (int t = 0; t < NT; ++t) {
        int   n  = n0b + t * 16 + lo;
        float bs = bias ? bias[n] : 0.0f;
        #pragma unroll
        for (int v = 0; v < 8; ++v) {
            int   m = m0 + v + hi8;           // C/D layout: M = vgpr + (lane/16)*8
            float c = acc[t][v] + bs;
            if (EPI == 2) c = (c > 0.0f) ? c : 0.01f * c;     // leaky_relu
            if (EPI == 3) c = 1.0f / (1.0f + __expf(-c));     // sigmoid
            if (EPI == 0 || EPI == 3)
                ((float*)Cout)[(size_t)m * N + n] = c;
            else
                ((bf16*)Cout)[(size_t)m * N + n] = (bf16)c;
        }
    }
}

// ---------------------------------------------------------------------------
// depthwise causal conv (width 4) + bias + SiLU; reads xc half of xz, emits bf16 u
// ---------------------------------------------------------------------------
__global__ void conv_silu_kernel(const float* __restrict__ xz,      // NTOK x 1024
                                 const float* __restrict__ conv_w,  // 512 x 4
                                 const float* __restrict__ conv_b,  // 512
                                 bf16* __restrict__ u)               // NTOK x 512
{
    int idx = blockIdx.x * blockDim.x + threadIdx.x;
    if (idx >= NTOK * D_INNER) return;
    int d = idx & (D_INNER - 1);
    int t = idx >> 9;            // token index b*L + l
    int l = t & (LL - 1);
    float acc = conv_b[d];
    #pragma unroll
    for (int k = 0; k < D_CONV; ++k) {
        int lk = l + k - (D_CONV - 1);
        if (lk >= 0)
            acc += xz[(size_t)(t + k - (D_CONV - 1)) * (2 * D_INNER) + d] * conv_w[d * D_CONV + k];
    }
    float s = acc / (1.0f + __expf(-acc));   // silu
    u[idx] = (bf16)s;
}

// ---------------------------------------------------------------------------
// selective scan: lane = (b, d); h[32] in registers; dbc row staged in LDS.
// Fuses: dt = softplus(dt_lo @ W_dt^T + b_dt), state update, y = h.C,
//        + u*D, * silu(z)  -> bf16 ygate for the W_out GEMM.
// ---------------------------------------------------------------------------
#define DCH 128
__global__ void scan_kernel(const float* __restrict__ dbc,    // NTOK x 80
                            const bf16*  __restrict__ u,      // NTOK x 512
                            const float* __restrict__ xz,     // NTOK x 1024 (z at col 512)
                            const float* __restrict__ W_dt,   // 512 x 16
                            const float* __restrict__ b_dt,   // 512
                            const float* __restrict__ A_log,  // 512 x 32
                            const float* __restrict__ Dp,     // 512
                            bf16* __restrict__ ygate)          // NTOK x 512
{
    const int b = blockIdx.x;
    const int d = blockIdx.y * DCH + threadIdx.x;
    __shared__ float sdbc[DT_RANK + 2 * D_STATE];   // 80 floats

    float wdt[DT_RANK];
    #pragma unroll
    for (int r = 0; r < DT_RANK; ++r) wdt[r] = W_dt[d * DT_RANK + r];
    float Arow[D_STATE];
    #pragma unroll
    for (int s = 0; s < D_STATE; ++s) Arow[s] = -__expf(A_log[d * D_STATE + s]);
    float h[D_STATE];
    #pragma unroll
    for (int s = 0; s < D_STATE; ++s) h[s] = 0.0f;

    const float bdt = b_dt[d], Dv = Dp[d];

    for (int l = 0; l < LL; ++l) {
        const size_t t = (size_t)b * LL + l;
        __syncthreads();
        if (threadIdx.x < DT_RANK + 2 * D_STATE)
            sdbc[threadIdx.x] = dbc[t * (DT_RANK + 2 * D_STATE) + threadIdx.x];
        __syncthreads();

        float dtr = bdt;
        #pragma unroll
        for (int r = 0; r < DT_RANK; ++r) dtr += sdbc[r] * wdt[r];
        float dt = (dtr > 20.0f) ? dtr : log1pf(__expf(dtr));   // softplus

        float uu  = (float)u[t * D_INNER + d];
        float dtu = dt * uu;
        float y   = 0.0f;
        #pragma unroll
        for (int s = 0; s < D_STATE; ++s) {
            h[s] = h[s] * __expf(dt * Arow[s]) + dtu * sdbc[DT_RANK + s];
            y   += h[s] * sdbc[DT_RANK + D_STATE + s];
        }
        float yv = y + uu * Dv;
        float z  = xz[t * (2 * D_INNER) + D_INNER + d];
        float g  = z / (1.0f + __expf(-z));                     // silu(z)
        ygate[t * D_INNER + d] = (bf16)(yv * g);
    }
}

// ---------------------------------------------------------------------------
extern "C" void kernel_launch(void* const* d_in, const int* in_sizes, int n_in,
                              void* d_out, int out_size, void* d_ws, size_t ws_size,
                              hipStream_t stream)
{
    const float* x      = (const float*)d_in[0];
    const float* W_in   = (const float*)d_in[1];
    const float* conv_w = (const float*)d_in[2];
    const float* conv_b = (const float*)d_in[3];
    const float* W_xproj= (const float*)d_in[4];
    const float* W_dt   = (const float*)d_in[5];
    const float* b_dt   = (const float*)d_in[6];
    const float* A_log  = (const float*)d_in[7];
    const float* Dp     = (const float*)d_in[8];
    const float* W_out  = (const float*)d_in[9];
    const float* W_ff1  = (const float*)d_in[10];
    const float* b_ff1  = (const float*)d_in[11];
    const float* W_ff2  = (const float*)d_in[12];
    const float* b_ff2  = (const float*)d_in[13];
    float* out = (float*)d_out;

    char* ws = (char*)d_ws;
    size_t o = 0;
    float* xz   = (float*)(ws + o); o += (size_t)NTOK * 1024 * 4;   // 134 MB
    float* dbc  = (float*)(ws + o); o += (size_t)NTOK * 80 * 4;     // 10.5 MB
    bf16*  xb   = (bf16*)(ws + o);  o += (size_t)NTOK * 256 * 2;    // 16.8 MB
    bf16*  ub   = (bf16*)(ws + o);  o += (size_t)NTOK * 512 * 2;    // 33.5 MB
    bf16*  ygb  = (bf16*)(ws + o);  o += (size_t)NTOK * 512 * 2;    // 33.5 MB
    bf16*  mb   = (bf16*)(ws + o);  o += (size_t)NTOK * 256 * 2;    // 16.8 MB
    bf16*  h2b  = (bf16*)(ws + o);  o += (size_t)NTOK * 32 * 2;     //  2.1 MB
    bf16*  Winb = (bf16*)(ws + o);  o += (size_t)1024 * 256 * 2;
    bf16*  Wxpb = (bf16*)(ws + o);  o += (size_t)80 * 512 * 2;
    bf16*  Woutb= (bf16*)(ws + o);  o += (size_t)256 * 512 * 2;
    bf16*  Wf1b = (bf16*)(ws + o);  o += (size_t)32 * 256 * 2;
    bf16*  Wf2b = (bf16*)(ws + o);  o += (size_t)256 * 32 * 2;

    // bf16 converts
    auto cvt = [&](const float* src, bf16* dst, int n) {
        cvt_bf16_kernel<<<(n + 255) / 256, 256, 0, stream>>>(src, dst, n);
    };
    cvt(x,      xb,   NTOK * 256);
    cvt(W_in,   Winb, 1024 * 256);
    cvt(W_xproj,Wxpb, 80 * 512);
    cvt(W_out,  Woutb,256 * 512);
    cvt(W_ff1,  Wf1b, 32 * 256);
    cvt(W_ff2,  Wf2b, 256 * 32);

    // G1: xz = x @ W_in^T          (32768 x 1024, K=256), f32 out.  N = 16*4*16
    wmma_gemm_kernel<0, 4><<<dim3(NTOK / 64, 16), 128, 0, stream>>>(
        xb, Winb, nullptr, xz, NTOK, 1024, 256);

    // conv4 + SiLU -> u (bf16)
    conv_silu_kernel<<<(NTOK * D_INNER) / 256, 256, 0, stream>>>(xz, conv_w, conv_b, ub);

    // G2: dbc = u @ W_xproj^T      (32768 x 80, K=512), f32 out.   N = 1*5*16
    wmma_gemm_kernel<0, 5><<<dim3(NTOK / 64, 1), 128, 0, stream>>>(
        ub, Wxpb, nullptr, dbc, NTOK, 80, 512);

    // selective scan -> gated y (bf16)
    scan_kernel<<<dim3(BB, D_INNER / DCH), DCH, 0, stream>>>(
        dbc, ub, xz, W_dt, b_dt, A_log, Dp, ygb);

    // G3: mamba = ygate @ W_out^T  (32768 x 256, K=512), bf16 out. N = 4*4*16
    wmma_gemm_kernel<1, 4><<<dim3(NTOK / 64, 4), 128, 0, stream>>>(
        ygb, Woutb, nullptr, mb, NTOK, 256, 512);

    // G4: h2 = leaky_relu(mamba @ W_ff1^T + b_ff1)  (32768 x 32, K=256). N = 1*2*16
    wmma_gemm_kernel<2, 2><<<dim3(NTOK / 64, 1), 128, 0, stream>>>(
        mb, Wf1b, b_ff1, h2b, NTOK, 32, 256);

    // G5: out = sigmoid(h2 @ W_ff2^T + b_ff2)       (32768 x 256, K=32). N = 4*4*16
    wmma_gemm_kernel<3, 4><<<dim3(NTOK / 64, 4), 128, 0, stream>>>(
        h2b, Wf2b, b_ff2, out, NTOK, 256, 32);
}
